// occo_11682311045323
// MI455X (gfx1250) — matrix operand order repaired and checked by reading, not compiled, not run on hardware
//
#include <hip/hip_runtime.h>
#include <math.h>

// ---------------------------------------------------------------------------
// OCCO dual-branch UNet forward for gfx1250 (MI455X).
// GEMM work runs through two WMMA kernels (strided GEMM + conv3x3 implicit
// im2col), both LDS-staged: per 32-K step the block stages A(32x32) and
// B(32x128) tiles into LDS as packed f16 pairs with coalesced f32 loads, then
// each wave reads contiguous-k fragments with 128-bit LDS loads and issues
// 4x v_wmma_f32_16x16x32_f16 (2x2 register blocking, 32x32 out tile / wave).
// Tensors are channel-major [C][B*H*W] ("CN layout").
// Param flattening assumed in make_params() insertion order after x,y.
// ---------------------------------------------------------------------------

typedef __attribute__((ext_vector_type(16))) _Float16 v16h;
typedef __attribute__((ext_vector_type(8)))  float    v8f;

__device__ __forceinline__ unsigned pk2(float a, float b) {
  union { _Float16 h[2]; unsigned u; } t;
  t.h[0] = (_Float16)a;
  t.h[1] = (_Float16)b;
  return t.u;
}

// A fragment (contiguous k order, matches ld_fragB): lane row, VGPR j -> kpair 8*half+j.
// ldsA is [row][16 kpairs].
__device__ __forceinline__ v16h ld_fragA(const unsigned* ldsA, int row, int half) {
  union { unsigned u[8]; v16h h; } t;
  const unsigned* p = ldsA + row * 16;
#pragma unroll
  for (int i = 0; i < 4; ++i) t.u[i] = p[8 * half + i];
#pragma unroll
  for (int i = 0; i < 4; ++i) t.u[4 + i] = p[8 * half + 4 + i];
  return t.h;
}

// B fragment (contiguous k order, matches ld_fragA): lane col, VGPR j -> kpair 8*half+j.
// ldsB is [col][16 kpairs] -> 8 consecutive dwords per lane (ds_load_b128 x2).
__device__ __forceinline__ v16h ld_fragB(const unsigned* ldsB, int col, int half) {
  union { unsigned u[8]; v16h h; } t;
  const unsigned* p = ldsB + col * 16 + 8 * half;
#pragma unroll
  for (int i = 0; i < 8; ++i) t.u[i] = p[i];
  return t.h;
}

__device__ __forceinline__ float gemm_epi(float v, int m,
                                          const float* bias, const float* scale,
                                          const float* shift, int act) {
  if (bias)  v += bias[m];
  if (scale) v = v * scale[m] + shift[m];
  if (act == 1)      v = fmaxf(v, 0.f);
  else if (act == 2) v = 1.f / (1.f + __expf(-v));
  return v;
}

// ---------------- strided GEMM: C[M,N] = epi(A x B) -------------------------
// A[m,k] = A[m*as_m + k*as_k]; B[k,n] = Bm[k*bs_k + n*bs_n]
__global__ void wmma_gemm_s(
    const float* __restrict__ A, long as_m, long as_k,
    const float* __restrict__ Bm, long bs_k, long bs_n,
    float* __restrict__ C, long cs_m, long cs_n,
    int M, int N, int K,
    const float* __restrict__ bias, const float* __restrict__ scale,
    const float* __restrict__ shift, int act)
{
  __shared__ unsigned ldsA[32 * 16];
  __shared__ unsigned ldsB[128 * 16];
  const int tid  = threadIdx.x;
  const int lane = tid & 31;
  const int wv   = tid >> 5;
  const int tM   = blockIdx.y << 5;
  const int tNb  = blockIdx.x << 7;
  const int mr   = lane & 15;
  const int half = lane >> 4;

  const int Mc = M - 1, Nc = N - 1;
  // staging assignments (clamped addresses; garbage rows/cols never stored)
  const int sAm  = tid >> 2;                 // tile row 0..31
  const int sAg  = (tid & 3) << 2;           // first kpair (0,4,8,12)
  const int gm   = tM + sAm;
  const float* Abase = A + (long)((gm < Mc) ? gm : Mc) * as_m;
  const int gn   = tNb + tid;                // tile col 0..127
  const float* Bbase = Bm + (long)((gn < Nc) ? gn : Nc) * bs_n;

  v8f a00 = {0.f,0.f,0.f,0.f,0.f,0.f,0.f,0.f};
  v8f a01 = a00, a10 = a00, a11 = a00;

  const int ksteps = (K + 31) >> 5;
  for (int ks = 0; ks < ksteps; ++ks) {
    const int k0 = ks << 5;
    __syncthreads();
#pragma unroll
    for (int i = 0; i < 4; ++i) {            // stage A: 4 kpairs / thread
      const int p = sAg + i;
      const int k = k0 + 2 * p;
      const float x0 = (k < K)     ? Abase[(long)k * as_k]       : 0.f;
      const float x1 = (k + 1 < K) ? Abase[(long)(k + 1) * as_k] : 0.f;
      ldsA[sAm * 16 + p] = pk2(x0, x1);
    }
#pragma unroll
    for (int p = 0; p < 16; ++p) {           // stage B: 16 kpairs / thread
      const int k = k0 + 2 * p;
      const float x0 = (k < K)     ? Bbase[(long)k * bs_k]       : 0.f;
      const float x1 = (k + 1 < K) ? Bbase[(long)(k + 1) * bs_k] : 0.f;
      ldsB[tid * 16 + p] = pk2(x0, x1);
    }
    __syncthreads();
    const v16h af0 = ld_fragA(ldsA, mr, half);
    const v16h af1 = ld_fragA(ldsA, mr + 16, half);
    const v16h bf0 = ld_fragB(ldsB, (wv << 5) + mr, half);
    const v16h bf1 = ld_fragB(ldsB, (wv << 5) + mr + 16, half);
    a00 = __builtin_amdgcn_wmma_f32_16x16x32_f16(false, af0, false, bf0, (short)0, a00, false, false);
    a01 = __builtin_amdgcn_wmma_f32_16x16x32_f16(false, af0, false, bf1, (short)0, a01, false, false);
    a10 = __builtin_amdgcn_wmma_f32_16x16x32_f16(false, af1, false, bf0, (short)0, a10, false, false);
    a11 = __builtin_amdgcn_wmma_f32_16x16x32_f16(false, af1, false, bf1, (short)0, a11, false, false);
  }

  const int nb0 = tNb + (wv << 5) + mr;
  const int nb1 = nb0 + 16;
  const bool n0ok = nb0 < N, n1ok = nb1 < N;
#pragma unroll
  for (int rr = 0; rr < 8; ++rr) {
    const int m0 = tM + rr + 8 * half;
    const int m1 = m0 + 16;
    if (m0 < M) {
      if (n0ok) C[(long)m0 * cs_m + (long)nb0 * cs_n] = gemm_epi(a00[rr], m0, bias, scale, shift, act);
      if (n1ok) C[(long)m0 * cs_m + (long)nb1 * cs_n] = gemm_epi(a01[rr], m0, bias, scale, shift, act);
    }
    if (m1 < M) {
      if (n0ok) C[(long)m1 * cs_m + (long)nb0 * cs_n] = gemm_epi(a10[rr], m1, bias, scale, shift, act);
      if (n1ok) C[(long)m1 * cs_m + (long)nb1 * cs_n] = gemm_epi(a11[rr], m1, bias, scale, shift, act);
    }
  }
}

// ---------- conv3x3 (pad 1) implicit-im2col GEMM, tap-major ----------------
// Weights A: [M][Cin*9] (element (m,ci,r) at m*Cin*9 + ci*9 + r).
// Act: channel-major [Cin][N], token n = b*HW + y*W + x.  C row stride cs_m,
// col stride 1.  K loop: 9 taps x Cin channels (staged 32 at a time).
__global__ void wmma_gemm_c3(
    const float* __restrict__ A,
    const float* __restrict__ Act, int Wimg, int Himg, int HWimg,
    float* __restrict__ C, long cs_m,
    int M, int N, int Cin,
    const float* __restrict__ bias, const float* __restrict__ scale,
    const float* __restrict__ shift, int act)
{
  __shared__ unsigned ldsA[32 * 16];
  __shared__ unsigned ldsB[128 * 16];
  const int tid  = threadIdx.x;
  const int lane = tid & 31;
  const int wv   = tid >> 5;
  const int tM   = blockIdx.y << 5;
  const int tNb  = blockIdx.x << 7;
  const int mr   = lane & 15;
  const int half = lane >> 4;

  const int Mc = M - 1, Nc = N - 1;
  const int sAm = tid >> 2;
  const int sAg = (tid & 3) << 2;
  const int gm  = tM + sAm;
  const long as_m = (long)Cin * 9;
  const float* AbaseM = A + (long)((gm < Mc) ? gm : Mc) * as_m;

  // per-thread staging column geometry (clamped)
  const int gn  = tNb + tid;
  const int ncl = (gn < Nc) ? gn : Nc;
  const int bi  = ncl / HWimg;
  const int prem = ncl - bi * HWimg;
  const int py  = prem / Wimg;
  const int px  = prem - py * Wimg;
  const float* ActImg = Act + (long)bi * HWimg;

  v8f a00 = {0.f,0.f,0.f,0.f,0.f,0.f,0.f,0.f};
  v8f a01 = a00, a10 = a00, a11 = a00;

  const int ksteps = (Cin + 31) >> 5;
  for (int r = 0; r < 9; ++r) {
    const int q3 = r / 3;
    const int dy = q3 - 1, dx = (r - q3 * 3) - 1;
    const int yy = py + dy, xx = px + dx;
    const bool vld = (unsigned)yy < (unsigned)Himg && (unsigned)xx < (unsigned)Wimg;
    const float* Bbase = vld ? (ActImg + (long)yy * Wimg + xx) : Act;
    const float msk = vld ? 1.f : 0.f;
    const float* Ar = AbaseM + r;

    for (int ks = 0; ks < ksteps; ++ks) {
      const int k0 = ks << 5;
      __syncthreads();
#pragma unroll
      for (int i = 0; i < 4; ++i) {          // stage A
        const int p = sAg + i;
        const int k = k0 + 2 * p;
        const float x0 = (k < Cin)     ? Ar[(long)k * 9]       : 0.f;
        const float x1 = (k + 1 < Cin) ? Ar[(long)(k + 1) * 9] : 0.f;
        ldsA[sAm * 16 + p] = pk2(x0, x1);
      }
#pragma unroll
      for (int p = 0; p < 16; ++p) {         // stage B (channel stride = N)
        const int k = k0 + 2 * p;
        const float x0 = (k < Cin)     ? Bbase[(long)k * N]       * msk : 0.f;
        const float x1 = (k + 1 < Cin) ? Bbase[(long)(k + 1) * N] * msk : 0.f;
        ldsB[tid * 16 + p] = pk2(x0, x1);
      }
      __syncthreads();
      const v16h af0 = ld_fragA(ldsA, mr, half);
      const v16h af1 = ld_fragA(ldsA, mr + 16, half);
      const v16h bf0 = ld_fragB(ldsB, (wv << 5) + mr, half);
      const v16h bf1 = ld_fragB(ldsB, (wv << 5) + mr + 16, half);
      a00 = __builtin_amdgcn_wmma_f32_16x16x32_f16(false, af0, false, bf0, (short)0, a00, false, false);
      a01 = __builtin_amdgcn_wmma_f32_16x16x32_f16(false, af0, false, bf1, (short)0, a01, false, false);
      a10 = __builtin_amdgcn_wmma_f32_16x16x32_f16(false, af1, false, bf0, (short)0, a10, false, false);
      a11 = __builtin_amdgcn_wmma_f32_16x16x32_f16(false, af1, false, bf1, (short)0, a11, false, false);
    }
  }

  const int nb0 = tNb + (wv << 5) + mr;
  const int nb1 = nb0 + 16;
  const bool n0ok = nb0 < N, n1ok = nb1 < N;
#pragma unroll
  for (int rr = 0; rr < 8; ++rr) {
    const int m0 = tM + rr + 8 * half;
    const int m1 = m0 + 16;
    if (m0 < M) {
      if (n0ok) C[(long)m0 * cs_m + nb0] = gemm_epi(a00[rr], m0, bias, scale, shift, act);
      if (n1ok) C[(long)m0 * cs_m + nb1] = gemm_epi(a01[rr], m0, bias, scale, shift, act);
    }
    if (m1 < M) {
      if (n0ok) C[(long)m1 * cs_m + nb0] = gemm_epi(a10[rr], m1, bias, scale, shift, act);
      if (n1ok) C[(long)m1 * cs_m + nb1] = gemm_epi(a11[rr], m1, bias, scale, shift, act);
    }
  }
}

// ------------------------- helper kernels ----------------------------------

__global__ void bn_prep_k(const float* __restrict__ g, const float* __restrict__ b,
                          const float* __restrict__ m, const float* __restrict__ v,
                          float* __restrict__ scale, float* __restrict__ shift, int C) {
  int i = blockIdx.x * 256 + threadIdx.x;
  if (i < C) {
    float s = g[i] * rsqrtf(v[i] + 1e-5f);
    scale[i] = s;
    shift[i] = b[i] - m[i] * s;
  }
}

__global__ void maxpool2_k(const float* __restrict__ in, float* __restrict__ out,
                           int C, int Bimg, int H, int W) {
  const int H2 = H >> 1, W2 = W >> 1;
  const long total = (long)C * Bimg * H2 * W2;
  long i = (long)blockIdx.x * 256 + threadIdx.x;
  if (i >= total) return;
  int x = (int)(i % W2); long t = i / W2;
  int y = (int)(t % H2); t /= H2;
  int b = (int)(t % Bimg);
  int c = (int)(t / Bimg);
  const float* p = in + ((long)c * Bimg + b) * (long)(H * W) + (long)(2 * y) * W + 2 * x;
  out[i] = fmaxf(fmaxf(p[0], p[1]), fmaxf(p[W], p[W + 1]));
}

__global__ void upsample2_k(const float* __restrict__ in, float* __restrict__ out,
                            int C, int Bimg, int H, int W) {
  const int Ho = 2 * H, Wo = 2 * W;
  const long total = (long)C * Bimg * Ho * Wo;
  long i = (long)blockIdx.x * 256 + threadIdx.x;
  if (i >= total) return;
  int x = (int)(i % Wo); long t = i / Wo;
  int y = (int)(t % Ho); t /= Ho;
  int b = (int)(t % Bimg);
  int c = (int)(t / Bimg);
  const float ry = (float)y * (float)(H - 1) / (float)(Ho - 1);
  const float rx = (float)x * (float)(W - 1) / (float)(Wo - 1);
  int y0 = (int)floorf(ry); int y1 = (y0 + 1 < H) ? y0 + 1 : H - 1; float wy = ry - y0;
  int x0 = (int)floorf(rx); int x1 = (x0 + 1 < W) ? x0 + 1 : W - 1; float wx = rx - x0;
  const float* p = in + ((long)c * Bimg + b) * (long)(H * W);
  float t0 = p[(long)y0 * W + x0] * (1.f - wx) + p[(long)y0 * W + x1] * wx;
  float t1 = p[(long)y1 * W + x0] * (1.f - wx) + p[(long)y1 * W + x1] * wx;
  out[i] = t0 * (1.f - wy) + t1 * wy;
}

__global__ void pool_avgmax_k(const float* __restrict__ X, float* __restrict__ yw,
                              int d, int Bimg, int HW, int t) {
  const int c = blockIdx.x, b = blockIdx.y;
  const float* p = X + ((long)c * Bimg + b) * (long)HW;
  float s = 0.f, mx = -3.0e38f;
  for (int i = threadIdx.x; i < HW; i += 256) { float v = p[i]; s += v; mx = fmaxf(mx, v); }
  __shared__ float ss[256];
  __shared__ float sm[256];
  ss[threadIdx.x] = s; sm[threadIdx.x] = mx; __syncthreads();
  for (int k = 128; k > 0; k >>= 1) {
    if ((int)threadIdx.x < k) {
      ss[threadIdx.x] += ss[threadIdx.x + k];
      sm[threadIdx.x] = fmaxf(sm[threadIdx.x], sm[threadIdx.x + k]);
    }
    __syncthreads();
  }
  if (threadIdx.x == 0) {
    yw[(long)b * 4 * d + (long)t * d + c]       = ss[0] / (float)HW;
    yw[(long)b * 4 * d + (long)(2 + t) * d + c] = sm[0];
  }
}

__global__ void small_linear_k(const float* __restrict__ in, const float* __restrict__ w,
                               const float* __restrict__ b, float* __restrict__ out,
                               int Bn, int I, int O, int act) {
  int idx = blockIdx.x * 256 + threadIdx.x;
  if (idx >= Bn * O) return;
  int o = idx % O, bb = idx / O;
  const float* wi = w + (long)o * I;
  const float* xi = in + (long)bb * I;
  float s = b ? b[o] : 0.f;
  for (int i = 0; i < I; ++i) s += wi[i] * xi[i];
  if (act == 1)      s = fmaxf(s, 0.f);
  else if (act == 2) s = 1.f / (1.f + __expf(-s));
  out[idx] = s;
}

__global__ void chan_mul_k(const float* __restrict__ x, const float* __restrict__ ywo,
                           float* __restrict__ out, int C, int Bimg, int HW,
                           int off, int rowlen) {
  const long total = (long)C * Bimg * HW;
  long i = (long)blockIdx.x * 256 + threadIdx.x;
  if (i >= total) return;
  long t = i / HW;
  int b = (int)(t % Bimg);
  int c = (int)(t / Bimg);
  out[i] = x[i] * ywo[(long)b * rowlen + off + c];
}

__global__ void spatial_mul_k(const float* __restrict__ x, const float* __restrict__ sw,
                              float* __restrict__ out, long Cn, long Nt) {
  long i = (long)blockIdx.x * 256 + threadIdx.x;
  if (i >= Cn) return;
  out[i] = x[i] * sw[i % Nt];
}

__global__ void softmax_d0_k(float* __restrict__ ctx, int BH, int dh, float scl) {
  int idx = blockIdx.x * 256 + threadIdx.x;
  if (idx >= BH * dh) return;
  int e = idx % dh;
  float* p = ctx + (long)(idx / dh) * dh * dh + e;
  float mx = -3.0e38f;
  for (int d0 = 0; d0 < dh; ++d0) mx = fmaxf(mx, p[(long)d0 * dh] * scl);
  float s = 0.f;
  for (int d0 = 0; d0 < dh; ++d0) {
    float v = __expf(p[(long)d0 * dh] * scl - mx);
    p[(long)d0 * dh] = v;
    s += v;
  }
  float inv = 1.f / s;
  for (int d0 = 0; d0 < dh; ++d0) p[(long)d0 * dh] *= inv;
}

__global__ void ln_add_k(const float* __restrict__ f, const float* __restrict__ e,
                         const float* __restrict__ g, const float* __restrict__ b,
                         float* __restrict__ out, int C, long Nt) {
  long n = (long)blockIdx.x * 256 + threadIdx.x;
  if (n >= Nt) return;
  float s1 = 0.f, s2 = 0.f;
  for (int c = 0; c < C; ++c) {
    float v = f[(long)c * Nt + n] + e[(long)c * Nt + n];
    s1 += v; s2 += v * v;
  }
  float mean = s1 / (float)C;
  float var  = fmaxf(s2 / (float)C - mean * mean, 0.f);
  float inv  = rsqrtf(var + 1e-5f);
  for (int c = 0; c < C; ++c) {
    float v = f[(long)c * Nt + n] + e[(long)c * Nt + n];
    out[(long)c * Nt + n] = (v - mean) * inv * g[c] + b[c];
  }
}

__global__ void dwconv3_k(const float* __restrict__ X, const float* __restrict__ Wt,
                          const float* __restrict__ bias, float* __restrict__ out,
                          int C, int Bimg, int H, int W) {
  const long total = (long)C * Bimg * H * W;
  long i = (long)blockIdx.x * 256 + threadIdx.x;
  if (i >= total) return;
  int x = (int)(i % W); long t = i / W;
  int y = (int)(t % H); t /= H;
  int b = (int)(t % Bimg);
  int c = (int)(t / Bimg);
  const float* p  = X + ((long)c * Bimg + b) * (long)(H * W);
  const float* wp = Wt + (long)c * 9;
  float s = bias[c];
#pragma unroll
  for (int dy = 0; dy < 3; ++dy) {
#pragma unroll
    for (int dx = 0; dx < 3; ++dx) {
      int yy = y + dy - 1, xx = x + dx - 1;
      if ((unsigned)yy < (unsigned)H && (unsigned)xx < (unsigned)W)
        s += p[(long)yy * W + xx] * wp[dy * 3 + dx];
    }
  }
  out[i] = fmaxf(s, 0.f);
}

__global__ void add_bn_k(const float* __restrict__ a, const float* __restrict__ b2,
                         const float* __restrict__ scale, const float* __restrict__ shift,
                         float* __restrict__ out, int C, long Nt) {
  const long total = (long)C * Nt;
  long i = (long)blockIdx.x * 256 + threadIdx.x;
  if (i >= total) return;
  int c = (int)(i / Nt);
  out[i] = (a[i] + b2[i]) * scale[c] + shift[c];
}

// ------------------------- host side ---------------------------------------

struct Convp { const float* w; const float* b; };
struct BNpar { const float *g, *b, *m, *v; };
struct DSpar { Convp c1; BNpar bn1; Convp c2; BNpar bn2; };
struct Linp  { const float* w; const float* b; };
struct LNpar { const float *g, *b; };
struct Fuspar {
  Linp l1, l2;
  Convp swc1, swc2;
  Linp p3, p4, kv1, kv2, e1, e2;
  LNpar ln1, ln2;
  Convp res, cec1, dw, cec2;
  BNpar cbn1, cbn2;
};
struct Uppar { Convp c1, c2; };

struct WS { char* p; size_t off; size_t cap; };
static float* wsal(WS& w, size_t nfloat) {
  size_t bytes = ((nfloat * 4) + 255) & ~(size_t)255;
  float* r = (float*)(w.p + w.off);
  w.off += bytes;
  return r;
}

static inline unsigned gblk(long total) { return (unsigned)((total + 255) / 256); }

static void gemmS(hipStream_t s,
                  const float* A, long am, long ak,
                  const float* Bm, long bk, long bn,
                  float* C, long cm, long cn,
                  int M, int N, int K,
                  const float* bias, const float* scale, const float* shift, int act) {
  dim3 grid((unsigned)((N + 127) / 128), (unsigned)((M + 31) / 32));
  wmma_gemm_s<<<grid, dim3(128), 0, s>>>(A, am, ak, Bm, bk, bn, C, cm, cn,
                                         M, N, K, bias, scale, shift, act);
}

static void gemmC3(hipStream_t s,
                   const float* Wt, const float* Act, int Wimg, int Himg,
                   float* C, long cm, int M, int N, int Cin,
                   const float* bias, const float* scale, const float* shift, int act) {
  dim3 grid((unsigned)((N + 127) / 128), (unsigned)((M + 31) / 32));
  wmma_gemm_c3<<<grid, dim3(128), 0, s>>>(Wt, Act, Wimg, Himg, Wimg * Himg,
                                          C, cm, M, N, Cin, bias, scale, shift, act);
}

static void run_bnprep(hipStream_t s, WS& ws, const BNpar& b, int C,
                       float** scale, float** shift) {
  *scale = wsal(ws, C);
  *shift = wsal(ws, C);
  bn_prep_k<<<gblk(C), 256, 0, s>>>(b.g, b.b, b.m, b.v, *scale, *shift, C);
}

static void run_ds(hipStream_t s, WS& ws, const DSpar& d, const float* in,
                   int Cin, int Cout, int Bimg, int H, int W,
                   float* X, float* Xp) {
  const long Nt = (long)Bimg * H * W;
  float *sc1, *sh1, *sc2, *sh2;
  run_bnprep(s, ws, d.bn1, Cout, &sc1, &sh1);
  run_bnprep(s, ws, d.bn2, Cout, &sc2, &sh2);
  float* T = wsal(ws, (size_t)Cout * Nt);
  gemmC3(s, d.c1.w, in, W, H, T, Nt, Cout, (int)Nt, Cin, d.c1.b, sc1, sh1, 1);
  gemmC3(s, d.c2.w, T, W, H, X, Nt, Cout, (int)Nt, Cout, d.c2.b, sc2, sh2, 1);
  if (Xp) {
    long tot = (long)Cout * Bimg * (H / 2) * (W / 2);
    maxpool2_k<<<gblk(tot), 256, 0, s>>>(X, Xp, Cout, Bimg, H, W);
  }
}

static void run_fusion(hipStream_t s, WS& ws, const Fuspar& F,
                       const float* X1, const float* X2,
                       int d, int heads, int Bimg, int H, int W, float* FU) {
  const long Nt = (long)Bimg * H * W;
  const int  HW = H * W;
  const int  dh = d / heads;

  // channel attention weights
  float* yw = wsal(ws, (size_t)Bimg * 4 * d);
  pool_avgmax_k<<<dim3(d, Bimg), 256, 0, s>>>(X1, yw, d, Bimg, HW, 0);
  pool_avgmax_k<<<dim3(d, Bimg), 256, 0, s>>>(X2, yw, d, Bimg, HW, 1);
  float* h1 = wsal(ws, (size_t)Bimg * 4 * d);
  small_linear_k<<<gblk(Bimg * 4 * d), 256, 0, s>>>(yw, F.l1.w, F.l1.b, h1, Bimg, 4 * d, 4 * d, 1);
  float* ywo = wsal(ws, (size_t)Bimg * 2 * d);
  small_linear_k<<<gblk(Bimg * 2 * d), 256, 0, s>>>(h1, F.l2.w, F.l2.b, ywo, Bimg, 4 * d, 2 * d, 2);

  // spatial gates (same sw1 weights for both inputs; faithful to reference)
  float* CAT1 = wsal(ws, (size_t)2 * d * Nt);
  float* CAT2 = wsal(ws, (size_t)2 * d * Nt);
  float* T    = wsal(ws, (size_t)d * Nt);
  float* swm  = wsal(ws, (size_t)Nt);
  gemmS(s, F.swc1.w, d, 1, X1, Nt, 1, T, Nt, 1, d, (int)Nt, d, F.swc1.b, nullptr, nullptr, 1);
  gemmS(s, F.swc2.w, d, 1, T, Nt, 1, swm, Nt, 1, 1, (int)Nt, d, F.swc2.b, nullptr, nullptr, 2);
  spatial_mul_k<<<gblk((long)d * Nt), 256, 0, s>>>(X1, swm, CAT1, (long)d * Nt, Nt);
  gemmS(s, F.swc1.w, d, 1, X2, Nt, 1, T, Nt, 1, d, (int)Nt, d, F.swc1.b, nullptr, nullptr, 1);
  gemmS(s, F.swc2.w, d, 1, T, Nt, 1, swm, Nt, 1, 1, (int)Nt, d, F.swc2.b, nullptr, nullptr, 2);
  spatial_mul_k<<<gblk((long)d * Nt), 256, 0, s>>>(X2, swm, CAT2, (long)d * Nt, Nt);

  // channel-weighted projections
  float* CW = wsal(ws, (size_t)d * Nt);
  float* Q1 = wsal(ws, (size_t)d * Nt);
  float* Q2 = wsal(ws, (size_t)d * Nt);
  chan_mul_k<<<gblk((long)d * Nt), 256, 0, s>>>(X1, ywo, CW, d, Bimg, HW, 0, 2 * d);
  gemmS(s, F.p3.w, d, 1, CW, Nt, 1, Q1, Nt, 1, d, (int)Nt, d, F.p3.b, nullptr, nullptr, 1);
  chan_mul_k<<<gblk((long)d * Nt), 256, 0, s>>>(X2, ywo, CW, d, Bimg, HW, d, 2 * d);
  gemmS(s, F.p4.w, d, 1, CW, Nt, 1, Q2, Nt, 1, d, (int)Nt, d, F.p4.b, nullptr, nullptr, 1);

  // kv projections
  float* KV1 = wsal(ws, (size_t)2 * d * Nt);
  float* KV2 = wsal(ws, (size_t)2 * d * Nt);
  gemmS(s, F.kv1.w, d, 1, Q1, Nt, 1, KV1, Nt, 1, 2 * d, (int)Nt, d, nullptr, nullptr, nullptr, 0);
  gemmS(s, F.kv2.w, d, 1, Q2, Nt, 1, KV2, Nt, 1, 2 * d, (int)Nt, d, nullptr, nullptr, nullptr, 0);

  // linear-attention contexts
  float* CTX1 = wsal(ws, (size_t)Bimg * heads * dh * dh);
  float* CTX2 = wsal(ws, (size_t)Bimg * heads * dh * dh);
  for (int b = 0; b < Bimg; ++b) {
    for (int h = 0; h < heads; ++h) {
      const float* k1 = KV1 + (long)(h * dh) * Nt + (long)b * HW;
      const float* v1 = KV1 + (long)(d + h * dh) * Nt + (long)b * HW;
      const float* k2 = KV2 + (long)(h * dh) * Nt + (long)b * HW;
      const float* v2 = KV2 + (long)(d + h * dh) * Nt + (long)b * HW;
      float* c1p = CTX1 + (long)(b * heads + h) * dh * dh;
      float* c2p = CTX2 + (long)(b * heads + h) * dh * dh;
      gemmS(s, k1, Nt, 1, v1, 1, Nt, c1p, dh, 1, dh, dh, HW, nullptr, nullptr, nullptr, 0);
      gemmS(s, k2, Nt, 1, v2, 1, Nt, c2p, dh, 1, dh, dh, HW, nullptr, nullptr, nullptr, 0);
    }
  }
  const float scl = 1.f / sqrtf((float)dh);
  softmax_d0_k<<<gblk(Bimg * heads * dh), 256, 0, s>>>(CTX1, Bimg * heads, dh, scl);
  softmax_d0_k<<<gblk(Bimg * heads * dh), 256, 0, s>>>(CTX2, Bimg * heads, dh, scl);

  // o1 = q1 @ ctx2 ; o2 = q2 @ ctx1
  for (int b = 0; b < Bimg; ++b) {
    for (int h = 0; h < heads; ++h) {
      const float* c1p = CTX1 + (long)(b * heads + h) * dh * dh;
      const float* c2p = CTX2 + (long)(b * heads + h) * dh * dh;
      const float* q1 = Q1 + (long)(h * dh) * Nt + (long)b * HW;
      const float* q2 = Q2 + (long)(h * dh) * Nt + (long)b * HW;
      float* o1 = CAT1 + (long)(d + h * dh) * Nt + (long)b * HW;
      float* o2 = CAT2 + (long)(d + h * dh) * Nt + (long)b * HW;
      gemmS(s, c2p, 1, dh, q1, Nt, 1, o1, Nt, 1, dh, HW, dh, nullptr, nullptr, nullptr, 0);
      gemmS(s, c1p, 1, dh, q2, Nt, 1, o2, Nt, 1, dh, HW, dh, nullptr, nullptr, nullptr, 0);
    }
  }

  // e1/e2 + residual layernorm into MCAT
  float* E    = wsal(ws, (size_t)d * Nt);
  float* MCAT = wsal(ws, (size_t)2 * d * Nt);
  gemmS(s, F.e1.w, 2 * d, 1, CAT1, Nt, 1, E, Nt, 1, d, (int)Nt, 2 * d, F.e1.b, nullptr, nullptr, 0);
  ln_add_k<<<gblk(Nt), 256, 0, s>>>(X1, E, F.ln1.g, F.ln1.b, MCAT, d, Nt);
  gemmS(s, F.e2.w, 2 * d, 1, CAT2, Nt, 1, E, Nt, 1, d, (int)Nt, 2 * d, F.e2.b, nullptr, nullptr, 0);
  ln_add_k<<<gblk(Nt), 256, 0, s>>>(X2, E, F.ln2.g, F.ln2.b, MCAT + (long)d * Nt, d, Nt);

  // conv-excite tail
  float *sc1, *sh1, *sc2, *sh2;
  run_bnprep(s, ws, F.cbn1, d, &sc1, &sh1);
  run_bnprep(s, ws, F.cbn2, d, &sc2, &sh2);
  float* RES = wsal(ws, (size_t)d * Nt);
  float* T2  = wsal(ws, (size_t)d * Nt);
  gemmS(s, F.res.w, 2 * d, 1, MCAT, Nt, 1, RES, Nt, 1, d, (int)Nt, 2 * d, nullptr, nullptr, nullptr, 0);
  gemmS(s, F.cec1.w, 2 * d, 1, MCAT, Nt, 1, T, Nt, 1, d, (int)Nt, 2 * d, F.cec1.b, nullptr, nullptr, 0);
  dwconv3_k<<<gblk((long)d * Nt), 256, 0, s>>>(T, F.dw.w, F.dw.b, T2, d, Bimg, H, W);
  gemmS(s, F.cec2.w, d, 1, T2, Nt, 1, T, Nt, 1, d, (int)Nt, d, F.cec2.b, sc1, sh1, 0);
  add_bn_k<<<gblk((long)d * Nt), 256, 0, s>>>(RES, T, sc2, sh2, FU, d, Nt);
}

static void run_up(hipStream_t s, WS& ws, const Uppar& U,
                   const float* a, int Ca, const float* bsrc, int Cb,
                   int Bimg, int Hb, int Wb, float* out, int Cout) {
  const int H = 2 * Hb, W = 2 * Wb;
  const long Nt = (long)Bimg * H * W;
  float* CAT = wsal(ws, (size_t)(Ca + Cb) * Nt);
  hipMemcpyAsync(CAT, a, (size_t)Ca * Nt * sizeof(float), hipMemcpyDeviceToDevice, s);
  upsample2_k<<<gblk((long)Cb * Nt), 256, 0, s>>>(bsrc, CAT + (long)Ca * Nt, Cb, Bimg, Hb, Wb);
  float* T = wsal(ws, (size_t)Cout * Nt);
  gemmC3(s, U.c1.w, CAT, W, H, T, Nt, Cout, (int)Nt, Ca + Cb, U.c1.b, nullptr, nullptr, 1);
  gemmC3(s, U.c2.w, T, W, H, out, Nt, Cout, (int)Nt, Cout, U.c2.b, nullptr, nullptr, 1);
}

// ------------------------- entry point -------------------------------------

extern "C" void kernel_launch(void* const* d_in, const int* in_sizes, int n_in,
                              void* d_out, int out_size, void* d_ws, size_t ws_size,
                              hipStream_t stream) {
  (void)in_sizes; (void)n_in; (void)out_size;
  const float* xin = (const float*)d_in[0];   // (2,1,256,256) == CN for C=1
  const float* yin = (const float*)d_in[1];

  int pi = 2;
  auto nxt = [&]() { return (const float*)d_in[pi++]; };
  auto getConv = [&](bool bias) { Convp c; c.w = nxt(); c.b = bias ? nxt() : nullptr; return c; };
  auto getBN   = [&]() { BNpar b; b.g = nxt(); b.b = nxt(); b.m = nxt(); b.v = nxt(); return b; };
  auto getDS   = [&]() { DSpar d; d.c1 = getConv(true); d.bn1 = getBN();
                         d.c2 = getConv(true); d.bn2 = getBN(); return d; };
  auto getLin  = [&](bool bias) { Linp l; l.w = nxt(); l.b = bias ? nxt() : nullptr; return l; };
  auto getLN   = [&]() { LNpar l; l.g = nxt(); l.b = nxt(); return l; };
  auto getFus  = [&]() {
    Fuspar f;
    f.l1 = getLin(true);  f.l2 = getLin(true);
    f.swc1 = getConv(true); f.swc2 = getConv(true);
    f.p3 = getLin(true);  f.p4 = getLin(true);
    f.kv1 = getLin(false); f.kv2 = getLin(false);
    f.e1 = getLin(true);  f.e2 = getLin(true);
    f.ln1 = getLN();      f.ln2 = getLN();
    f.res = getConv(false); f.cec1 = getConv(true);
    f.dw = getConv(true);   f.cec2 = getConv(true);
    f.cbn1 = getBN();       f.cbn2 = getBN();
    return f;
  };
  auto getUp = [&]() { Uppar u; u.c1 = getConv(true); u.c2 = getConv(true); return u; };

  DSpar v1 = getDS(), v2 = getDS(), v3 = getDS(), v4 = getDS();
  DSpar i1 = getDS(), i2 = getDS(), i3 = getDS(), i4 = getDS();
  Fuspar f1 = getFus(), f2 = getFus(), f3 = getFus(), f4 = getFus();
  Uppar up3 = getUp(), up2 = getUp(), up1 = getUp();
  Convp fin = getConv(true);

  WS ws; ws.p = (char*)d_ws; ws.off = 0; ws.cap = ws_size;
  const int B = 2;
  float* X1 = wsal(ws, (size_t)32 * 131072);  float* X1p = wsal(ws, (size_t)32 * 32768);
  float* X2 = wsal(ws, (size_t)64 * 32768);   float* X2p = wsal(ws, (size_t)64 * 8192);
  float* X3 = wsal(ws, (size_t)128 * 8192);   float* X3p = wsal(ws, (size_t)128 * 2048);
  float* X4 = wsal(ws, (size_t)256 * 2048);
  float* Y1 = wsal(ws, (size_t)32 * 131072);  float* Y1p = wsal(ws, (size_t)32 * 32768);
  float* Y2 = wsal(ws, (size_t)64 * 32768);   float* Y2p = wsal(ws, (size_t)64 * 8192);
  float* Y3 = wsal(ws, (size_t)128 * 8192);   float* Y3p = wsal(ws, (size_t)128 * 2048);
  float* Y4 = wsal(ws, (size_t)256 * 2048);
  float* FU1 = wsal(ws, (size_t)32 * 131072);
  float* FU2 = wsal(ws, (size_t)64 * 32768);
  float* FU3 = wsal(ws, (size_t)128 * 8192);
  float* FU4 = wsal(ws, (size_t)256 * 2048);
  float* U3 = wsal(ws, (size_t)128 * 8192);
  float* U2 = wsal(ws, (size_t)64 * 32768);
  float* U1 = wsal(ws, (size_t)32 * 131072);
  const size_t mark = ws.off;

  run_ds(stream, ws, v1, xin, 1,  32, B, 256, 256, X1, X1p); ws.off = mark;
  run_ds(stream, ws, v2, X1p, 32, 64, B, 128, 128, X2, X2p); ws.off = mark;
  run_ds(stream, ws, v3, X2p, 64, 128, B, 64, 64, X3, X3p);  ws.off = mark;
  run_ds(stream, ws, v4, X3p, 128, 256, B, 32, 32, X4, nullptr); ws.off = mark;
  run_ds(stream, ws, i1, yin, 1,  32, B, 256, 256, Y1, Y1p); ws.off = mark;
  run_ds(stream, ws, i2, Y1p, 32, 64, B, 128, 128, Y2, Y2p); ws.off = mark;
  run_ds(stream, ws, i3, Y2p, 64, 128, B, 64, 64, Y3, Y3p);  ws.off = mark;
  run_ds(stream, ws, i4, Y3p, 128, 256, B, 32, 32, Y4, nullptr); ws.off = mark;

  run_fusion(stream, ws, f1, X1, Y1, 32,  4, B, 256, 256, FU1); ws.off = mark;
  run_fusion(stream, ws, f2, X2, Y2, 64,  4, B, 128, 128, FU2); ws.off = mark;
  run_fusion(stream, ws, f3, X3, Y3, 128, 4, B, 64, 64, FU3);   ws.off = mark;
  run_fusion(stream, ws, f4, X4, Y4, 256, 1, B, 32, 32, FU4);   ws.off = mark;

  run_up(stream, ws, up3, FU3, 128, FU4, 256, B, 32, 32, U3, 128); ws.off = mark;
  run_up(stream, ws, up2, FU2, 64, U3, 128, B, 64, 64, U2, 64);    ws.off = mark;
  run_up(stream, ws, up1, FU1, 32, U2, 64, B, 128, 128, U1, 32);   ws.off = mark;

  gemmS(stream, fin.w, 32, 1, U1, 131072, 1,
        (float*)d_out, 131072, 1, 1, 131072, 32, fin.b, nullptr, nullptr, 0);
}
